// STDAMHGN_69672959476360
// MI455X (gfx1250) — matrix operand order repaired and checked by reading, not compiled, not run on hardware
//
#include <hip/hip_runtime.h>
#include <math.h>

#define DEV __device__ __forceinline__

typedef __attribute__((ext_vector_type(16))) __bf16 v16bf;
typedef __attribute__((ext_vector_type(8)))  float  v8f;

constexpr int B_ = 4, T_ = 12, V_ = 10000, D_ = 64, H_ = 3, E_ = 512, S_ = 32;
constexpr int N_      = B_ * T_;   // 48 graph "frames"
constexpr int ROWS_NV = N_ * V_;   // 480000 rows of (frame, vertex)
constexpr int ROWS_BV = B_ * V_;   // 40000 LSTM sequences

// ---------------- bf16 helpers (native hardware converts) ----------------
DEV unsigned short f2bf(float x) {
  union { __bf16 b; unsigned short s; } u;
  u.b = (__bf16)x;                       // v_cvt_pk_bf16_f32 path (RNE)
  return u.s;
}
DEV float bf2f(unsigned short s) {
  union { unsigned short s; __bf16 b; } u;
  u.s = s;
  return (float)u.b;
}
// 16 contiguous bf16 (32B) -> v16bf (B operand / contiguous A chunks)
DEV v16bf load16(const unsigned short* p) {
  union { uint4 q[2]; v16bf v; } u;
  u.q[0] = *(const uint4*)(p);
  u.q[1] = *(const uint4*)(p + 8);
  return u.v;
}
// A operand: two 16B chunks (lo: K=32ks+16h.., hi: K=32ks+16h+8..)
DEV v16bf loadA(const unsigned short* lo, const unsigned short* hi) {
  union { uint4 q[2]; v16bf v; } u;
  u.q[0] = *(const uint4*)(lo);
  u.q[1] = *(const uint4*)(hi);
  return u.v;
}
DEV v8f wmma_bf16(v16bf a, v16bf b, v8f c) {
  return __builtin_amdgcn_wmma_f32_16x16x32_bf16(false, a, false, b,
                                                 (short)0, c, false, false);
}
DEV float sigm(float x) { return 1.0f / (1.0f + expf(-x)); }

// ---------------- weight prep: bf16 conversions + attention collapse ----------------
__global__ void prep_kernel(const float* w2, const float* wih, const float* whh,
                            const float* bih, const float* bhh,
                            const float* attw, const float* atta,
                            unsigned short* w2T, unsigned short* wihB,
                            unsigned short* whhB, float* biasG, float* uv) {
  int idx = blockIdx.x * blockDim.x + threadIdx.x;
  if (idx < 4096) {                       // w2T[n][k] = w2[k][n]
    int k = idx >> 6, n = idx & 63;
    w2T[n * 64 + k] = f2bf(w2[k * 64 + n]);
  } else if (idx < 4096 + 16384) {
    int i = idx - 4096; wihB[i] = f2bf(wih[i]);
  } else if (idx < 4096 + 32768) {
    int i = idx - 4096 - 16384; whhB[i] = f2bf(whh[i]);
  } else if (idx < 4096 + 32768 + 256) {
    int i = idx - 4096 - 32768; biasG[i] = bih[i] + bhh[i];
  } else if (idx < 4096 + 32768 + 256 + 128) {
    int i = idx - (4096 + 32768 + 256);
    int r = i & 63, hi = i >> 6;          // uv[0..63]=attn_w@a_lo, uv[64..127]=attn_w@a_hi
    float s = 0.f;
    for (int j = 0; j < 64; ++j) s += attw[r * 64 + j] * atta[hi * 64 + j];
    uv[hi * 64 + r] = s;
  }
}

// ---------------- degree ----------------
__global__ void deg_kernel(const int* edges, float* deg) {
  int idx = blockIdx.x * blockDim.x + threadIdx.x;
  if (idx >= H_ * E_ * S_) return;
  int h = idx / (E_ * S_);
  atomicAdd(&deg[h * V_ + edges[idx]], 1.0f);
}

// ---------------- hypersage pass 1 (f = 1 -> D) ----------------
__global__ void hs1_gather(const float* seq, const int* edges_h, float* agg1) {
  int idx = blockIdx.x * blockDim.x + threadIdx.x;    // n*E + e
  if (idx >= N_ * E_) return;
  int n = idx >> 9, e = idx & (E_ - 1);
  const int* ep = edges_h + e * S_;
  float s = 0.f;
  #pragma unroll 8
  for (int j = 0; j < S_; ++j) s += seq[n * V_ + ep[j]];
  agg1[idx] = s * (1.0f / S_);
}
__global__ void hs1_scatter(const float* agg1, const int* edges_h, float* Xo) {
  int idx = blockIdx.x * blockDim.x + threadIdx.x;    // n*E*S
  if (idx >= N_ * E_ * S_) return;
  int n = idx / (E_ * S_);
  int r = idx - n * (E_ * S_);
  int e = r / S_, s = r - e * S_;
  int v = edges_h[e * S_ + s];
  atomicAdd(&Xo[n * V_ + v], agg1[n * E_ + e]);
}
__global__ void hs1_out(const float* Xo, const float* deg_h, const float* w1,
                        const float* b1, unsigned short* Y1) {
  int idx = blockIdx.x * blockDim.x + threadIdx.x;    // rows*64
  if (idx >= ROWS_NV * 64) return;
  int row = idx >> 6, d = idx & 63;
  int v = row % V_;
  float x = Xo[row] / fmaxf(deg_h[v], 1.0f);
  Y1[idx] = f2bf(x * w1[d] + b1[d]);
}

// ---------------- hypersage pass 2 (f = D), aggregation ----------------
__global__ void hs2_gather(const unsigned short* Y1, const int* edges_h, float* agg2) {
  int idx = blockIdx.x * blockDim.x + threadIdx.x;    // n*E*64
  if (idx >= N_ * E_ * 64) return;
  int n = idx / (E_ * 64);
  int r = idx - n * (E_ * 64);
  int e = r >> 6, d = r & 63;
  const int* ep = edges_h + e * S_;
  float s = 0.f;
  #pragma unroll 8
  for (int j = 0; j < S_; ++j) s += bf2f(Y1[(size_t)(n * V_ + ep[j]) * 64 + d]);
  agg2[idx] = s * (1.0f / S_);
}
__global__ void hs2_scatter(const float* agg2, const int* edges_h, float* Z) {
  int idx = blockIdx.x * blockDim.x + threadIdx.x;    // n*E*S*64
  if (idx >= N_ * E_ * S_ * 64) return;
  int d = idx & 63;
  int s = (idx >> 6) & (S_ - 1);
  int e = (idx >> 11) & (E_ - 1);
  int n = idx >> 20;
  int v = edges_h[e * S_ + s];
  atomicAdd(&Z[(size_t)(n * V_ + v) * 64 + d], agg2[(size_t)(n * E_ + e) * 64 + d]);
}

// build bf16 A-operand from f32 row with per-row scale (deg normalize fused)
DEV v16bf a_from_f32(const float* row, int ks, int half, float scale) {
  union { __bf16 b[16]; v16bf v; } u;
  int klo = 32 * ks + 16 * half;
  int khi = 32 * ks + 16 * half + 8;
  #pragma unroll
  for (int i = 0; i < 8; ++i) {
    u.b[i]     = (__bf16)(row[klo + i] * scale);
    u.b[8 + i] = (__bf16)(row[khi + i] * scale);
  }
  return u.v;
}

// ---------------- WMMA GEMM: xs = (Z/deg) @ w2 + b2  (480k x 64 x 64) ----------------
__global__ void gemm_xs(const float* Z, const float* deg_h, const unsigned short* w2T,
                        const float* b2, unsigned short* xs) {
  int wave = threadIdx.x >> 5, lane = threadIdx.x & 31;
  int tile = blockIdx.x * 4 + wave;                   // 30000 tiles exactly
  int row0 = tile * 16;
  int m = lane & 15, half = lane >> 4;
  int rowm = row0 + m;
  float scale = 1.0f / fmaxf(deg_h[rowm % V_], 1.0f);
  const float* zr = Z + (size_t)rowm * 64;
  v16bf a0 = a_from_f32(zr, 0, half, scale);
  v16bf a1 = a_from_f32(zr, 1, half, scale);
  int n = lane & 15;
  #pragma unroll
  for (int nt = 0; nt < 4; ++nt) {
    int col = nt * 16 + n;
    float bv = b2[col];
    v8f acc = {bv, bv, bv, bv, bv, bv, bv, bv};
    acc = wmma_bf16(a0, load16(w2T + col * 64 + 16 * half), acc);
    acc = wmma_bf16(a1, load16(w2T + col * 64 + 32 + 16 * half), acc);
    #pragma unroll
    for (int j = 0; j < 8; ++j)
      xs[(size_t)(row0 + j + 8 * half) * 64 + col] = f2bf(acc[j]);
  }
}

// ---------------- attention (collapsed) + transpose to LSTM layout ----------------
__global__ void attn_kernel(const unsigned short* xs0, const unsigned short* xs1,
                            const unsigned short* xs2, const float* uv,
                            unsigned short* fused_t) {
  int R = blockIdx.x * blockDim.x + threadIdx.x;
  if (R >= ROWS_NV) return;
  int nn = R / V_, v = R - nn * V_;
  int b = nn / T_, t = nn - b * T_;
  const unsigned short* p0 = xs0 + (size_t)R * 64;
  const unsigned short* p1 = xs1 + (size_t)R * 64;
  const unsigned short* p2 = xs2 + (size_t)R * 64;
  float s0 = 0.f, s1 = 0.f, s2 = 0.f, sr = 0.f;
  #pragma unroll 8
  for (int d = 0; d < 64; ++d) {
    float x0 = bf2f(p0[d]), x1 = bf2f(p1[d]), x2 = bf2f(p2[d]);
    float ud = uv[d];
    s0 += x0 * ud; s1 += x1 * ud; s2 += x2 * ud;
    sr += (x0 + x1 + x2) * (1.0f / 3.0f) * uv[64 + d];
  }
  float z0 = s0 + sr, z1 = s1 + sr, z2 = s2 + sr;
  z0 = z0 > 0.f ? z0 : 0.2f * z0;
  z1 = z1 > 0.f ? z1 : 0.2f * z1;
  z2 = z2 > 0.f ? z2 : 0.2f * z2;
  float mx = fmaxf(z0, fmaxf(z1, z2));
  float e0 = expf(z0 - mx), e1 = expf(z1 - mx), e2 = expf(z2 - mx);
  float inv = 1.0f / (e0 + e1 + e2);
  float a0 = e0 * inv, a1 = e1 * inv, a2 = e2 * inv;
  unsigned short* o = fused_t + ((size_t)(b * V_ + v) * T_ + t) * 64;
  #pragma unroll 8
  for (int d = 0; d < 64; ++d)
    o[d] = f2bf(a0 * bf2f(p0[d]) + a1 * bf2f(p1[d]) + a2 * bf2f(p2[d]));
}

// ---------------- LSTM: 16 rows per wave, WMMA gates, h in LDS (bf16) ----------------
__global__ void lstm_kernel(const unsigned short* fused_t, const unsigned short* wih,
                            const unsigned short* whh, const float* biasG,
                            float* hbuf) {
  __shared__ unsigned short sH[4][2][16 * 64];        // per-wave double-buffered h
  int wave = threadIdx.x >> 5, lane = threadIdx.x & 31;
  int tile = blockIdx.x * 4 + wave;                   // 2500 tiles exactly
  int row0 = tile * 16;
  int m = lane & 15, half = lane >> 4, n = lane & 15;

  for (int k = lane; k < 1024; k += 32) sH[wave][0][k] = 0;
  float cst[4][8];
  #pragma unroll
  for (int a = 0; a < 4; ++a)
    #pragma unroll
    for (int j = 0; j < 8; ++j) cst[a][j] = 0.f;
  asm volatile("s_wait_dscnt 0x0" ::: "memory");

  int buf = 0;
  #pragma unroll 1
  for (int t = 0; t < T_; ++t) {
    const unsigned short* xr = fused_t + ((size_t)(row0 + m) * T_ + t) * 64;
    v16bf ax0 = loadA(xr + 16 * half, xr + 16 * half + 8);
    v16bf ax1 = loadA(xr + 32 + 16 * half, xr + 40 + 16 * half);
    if (t + 1 < T_)                                   // warm next x-tile into WGP$
      __builtin_prefetch(xr + 64, 0, 1);
    const unsigned short* hr = &sH[wave][buf][m * 64];
    v16bf ah0 = loadA(hr + 16 * half, hr + 16 * half + 8);
    v16bf ah1 = loadA(hr + 32 + 16 * half, hr + 40 + 16 * half);

    #pragma unroll
    for (int nt = 0; nt < 4; ++nt) {
      v8f acc[4];
      #pragma unroll
      for (int g = 0; g < 4; ++g) {                   // i, f, g, o
        int colg = g * 64 + nt * 16 + n;
        float bv = biasG[colg];
        acc[g] = (v8f){bv, bv, bv, bv, bv, bv, bv, bv};
        const unsigned short* wi = wih + (size_t)colg * 64;
        const unsigned short* wh = whh + (size_t)colg * 64;
        acc[g] = wmma_bf16(ax0, load16(wi + 16 * half), acc[g]);
        acc[g] = wmma_bf16(ax1, load16(wi + 32 + 16 * half), acc[g]);
        acc[g] = wmma_bf16(ah0, load16(wh + 16 * half), acc[g]);
        acc[g] = wmma_bf16(ah1, load16(wh + 32 + 16 * half), acc[g]);
      }
      #pragma unroll
      for (int j = 0; j < 8; ++j) {
        float ig = sigm(acc[0][j]);
        float fg = sigm(acc[1][j]);
        float gg = tanhf(acc[2][j]);
        float og = sigm(acc[3][j]);
        float cn = fg * cst[nt][j] + ig * gg;
        cst[nt][j] = cn;
        float hn = og * tanhf(cn);
        int mm = j + 8 * half;
        int col = nt * 16 + n;
        sH[wave][buf ^ 1][mm * 64 + col] = f2bf(hn);
        if (t == T_ - 1) hbuf[(size_t)(row0 + mm) * 64 + col] = hn;
      }
    }
    asm volatile("s_wait_dscnt 0x0" ::: "memory");
    buf ^= 1;
  }
}

// ---------------- output head ----------------
__global__ void out_kernel(const float* h0, const float* h1, const float* w_out,
                           const float* b_out, float* out) {
  int r = blockIdx.x * blockDim.x + threadIdx.x;
  if (r >= ROWS_BV) return;
  float s = b_out[0];
  #pragma unroll 8
  for (int d = 0; d < 64; ++d)
    s += h0[(size_t)r * 64 + d] * w_out[d] + h1[(size_t)r * 64 + d] * w_out[64 + d];
  out[r] = s;
}

// ---------------- host ----------------
extern "C" void kernel_launch(void* const* d_in, const int* in_sizes, int n_in,
                              void* d_out, int out_size, void* d_ws, size_t ws_size,
                              hipStream_t stream) {
  const float* tendency    = (const float*)d_in[0];
  const float* periodicity = (const float*)d_in[1];
  const int*   edges       = (const int*)d_in[2];
  const float* w1     = (const float*)d_in[3];
  const float* b1     = (const float*)d_in[4];
  const float* w2     = (const float*)d_in[5];
  const float* b2     = (const float*)d_in[6];
  const float* attn_w = (const float*)d_in[7];
  const float* attn_a = (const float*)d_in[8];
  const float* w_ih   = (const float*)d_in[9];
  const float* w_hh   = (const float*)d_in[10];
  const float* b_ih   = (const float*)d_in[11];
  const float* b_hh   = (const float*)d_in[12];
  const float* w_out  = (const float*)d_in[13];
  const float* b_out  = (const float*)d_in[14];
  float* out = (float*)d_out;

  char* base = (char*)d_ws;
  size_t off = 0;
  auto alloc = [&](size_t bytes) -> void* {
    off = (off + 255) & ~size_t(255);
    void* p = base + off;
    off += bytes;
    return p;
  };
  float*          deg    = (float*)alloc((size_t)H_ * V_ * 4);
  unsigned short* w2T    = (unsigned short*)alloc(4096 * 2);
  unsigned short* wihB   = (unsigned short*)alloc(16384 * 2);
  unsigned short* whhB   = (unsigned short*)alloc(16384 * 2);
  float*          biasG  = (float*)alloc(256 * 4);
  float*          uv     = (float*)alloc(128 * 4);
  float*          agg1   = (float*)alloc((size_t)N_ * E_ * 4);
  float*          Xo     = (float*)alloc((size_t)ROWS_NV * 4);
  unsigned short* Y1     = (unsigned short*)alloc((size_t)ROWS_NV * 64 * 2);
  float*          agg2   = (float*)alloc((size_t)N_ * E_ * 64 * 4);
  float*          Z      = (float*)alloc((size_t)ROWS_NV * 64 * 4);
  unsigned short* xsb[3];
  for (int h = 0; h < 3; ++h)
    xsb[h] = (unsigned short*)alloc((size_t)ROWS_NV * 64 * 2);
  unsigned short* fusedT = (unsigned short*)alloc((size_t)ROWS_BV * T_ * 64 * 2);
  float*          hbuf0  = (float*)alloc((size_t)ROWS_BV * 64 * 4);
  float*          hbuf1  = (float*)alloc((size_t)ROWS_BV * 64 * 4);

  // degrees + weight prep (once)
  hipMemsetAsync(deg, 0, (size_t)H_ * V_ * 4, stream);
  deg_kernel<<<(H_ * E_ * S_ + 255) / 256, 256, 0, stream>>>(edges, deg);
  prep_kernel<<<(37632 + 255) / 256, 256, 0, stream>>>(
      w2, w_ih, w_hh, b_ih, b_hh, attn_w, attn_a, w2T, wihB, whhB, biasG, uv);

  for (int br = 0; br < 2; ++br) {
    const float* seq = (br == 0) ? tendency : periodicity;
    float* hb = (br == 0) ? hbuf0 : hbuf1;
    for (int h = 0; h < H_; ++h) {
      const int* eh = edges + (size_t)h * E_ * S_;
      const float* dh = deg + (size_t)h * V_;
      hs1_gather<<<(N_ * E_ + 255) / 256, 256, 0, stream>>>(seq, eh, agg1);
      hipMemsetAsync(Xo, 0, (size_t)ROWS_NV * 4, stream);
      hs1_scatter<<<(N_ * E_ * S_ + 255) / 256, 256, 0, stream>>>(agg1, eh, Xo);
      hs1_out<<<(ROWS_NV * 64 + 255) / 256, 256, 0, stream>>>(Xo, dh, w1, b1, Y1);
      hs2_gather<<<(N_ * E_ * 64 + 255) / 256, 256, 0, stream>>>(Y1, eh, agg2);
      hipMemsetAsync(Z, 0, (size_t)ROWS_NV * 64 * 4, stream);
      hs2_scatter<<<(N_ * E_ * S_ * 64 + 255) / 256, 256, 0, stream>>>(agg2, eh, Z);
      gemm_xs<<<ROWS_NV / 16 / 4, 128, 0, stream>>>(Z, dh, w2T, b2, xsb[h]);
    }
    attn_kernel<<<(ROWS_NV + 255) / 256, 256, 0, stream>>>(
        xsb[0], xsb[1], xsb[2], uv, fusedT);
    lstm_kernel<<<ROWS_BV / 16 / 4, 128, 0, stream>>>(fusedT, wihB, whhB, biasG, hb);
  }
  out_kernel<<<(ROWS_BV + 255) / 256, 256, 0, stream>>>(hbuf0, hbuf1, w_out, b_out, out);
}
